// CSALightningIndexer_40286793236979
// MI455X (gfx1250) — matrix operand order repaired and checked by reading, not compiled, not run on hardware
//
#include <hip/hip_runtime.h>
#include <hip/hip_bf16.h>

// Problem constants (match reference)
#define BATCH   2
#define LQ      4096
#define LC      1024
#define DMODEL  2048
#define NHEADS  4
#define HEADDIM 512

typedef __attribute__((ext_vector_type(16))) __bf16 v16bf;
typedef __attribute__((ext_vector_type(8)))  __bf16 v8bf;
typedef __attribute__((ext_vector_type(8)))  float  v8f;

#define WMMA_BF16(A, B, C) \
    __builtin_amdgcn_wmma_f32_16x16x32_bf16(false, (A), false, (B), (short)0, (C), false, false)

#define SHUF16(lo, hi) \
    __builtin_shufflevector((lo), (hi), 0, 1, 2, 3, 4, 5, 6, 7, 8, 9, 10, 11, 12, 13, 14, 15)

// ---------------------------------------------------------------------------
// A 16x32 bf16 fragment (ISA 7.12.2): lane L holds row M = L%16; K halves
// 0..7 = g*8+0..7, halves 8..15 = 16+g*8+0..7 with g = L/16.
// Source row-major [*, lda], K contiguous -> two 16-byte loads per lane.
// ---------------------------------------------------------------------------
__device__ __forceinline__ v16bf load_a_frag_f32(const float* __restrict__ A, int lda,
                                                 int mbase, int kbase, int lane) {
    int m = mbase + (lane & 15);
    int g = (lane >> 4) * 8;
    const float* p0 = A + (size_t)m * lda + kbase + g;
    const float* p1 = p0 + 16;
    v16bf a;
#pragma unroll
    for (int i = 0; i < 8; ++i) a[i]     = (__bf16)p0[i];   // merges into b128 + cvt_pk
#pragma unroll
    for (int i = 0; i < 8; ++i) a[8 + i] = (__bf16)p1[i];
    return a;
}

__device__ __forceinline__ v16bf load_a_frag_bf(const __bf16* __restrict__ A, int lda,
                                                int mbase, int kbase, int lane) {
    int m = mbase + (lane & 15);
    int g = (lane >> 4) * 8;
    const __bf16* p = A + (size_t)m * lda + kbase + g;
    v8bf lo = *(const v8bf*)p;
    v8bf hi = *(const v8bf*)(p + 16);
    return SHUF16(lo, hi);
}

// ---------------------------------------------------------------------------
// B 32x16 fragment from an LDS slab laid out [64 n-rows][32 k] bf16.
// Lane L: column n = t*16 + L%16, K = (L/16)*8 + 0..7 and 16 + (L/16)*8 + 0..7.
// ---------------------------------------------------------------------------
__device__ __forceinline__ v16bf load_b_frag_lds(const __bf16* slab, int t, int lane) {
    const __bf16* p = slab + ((t * 16 + (lane & 15)) << 5) + ((lane >> 4) << 3);
    v8bf lo = *(const v8bf*)p;
    v8bf hi = *(const v8bf*)(p + 16);
    return SHUF16(lo, hi);
}

// Stage 64 rows x 32 k of a row-major fp32 matrix into LDS as bf16.
// 256 threads, each moves 8 elements (32B global load -> cvt -> 16B LDS store).
__device__ __forceinline__ void stage_slab_f32(const float* __restrict__ src, int lds,
                                               int nbase, int kpos, __bf16* dst, int tid) {
    int row   = tid >> 2;          // 0..63
    int chunk = (tid & 3) * 8;     // 0,8,16,24
    const float* p = src + (size_t)(nbase + row) * lds + kpos + chunk;
    v8bf v;
#pragma unroll
    for (int i = 0; i < 8; ++i) v[i] = (__bf16)p[i];
    *(v8bf*)(dst + row * 32 + chunk) = v;
}

// Same but bf16 source (straight 16B global_load_b128 -> ds_store_b128 copy).
__device__ __forceinline__ void stage_slab_bf(const __bf16* __restrict__ src, int lds,
                                              int nbase, int kpos, __bf16* dst, int tid) {
    int row   = tid >> 2;
    int chunk = (tid & 3) * 8;
    *(v8bf*)(dst + row * 32 + chunk) =
        *(const v8bf*)(src + (size_t)(nbase + row) * lds + kpos + chunk);
}

// ---------------------------------------------------------------------------
// Stage 1: C_bf16[M,N] = A[M,K] * W[N,K]^T  (q/k projections)
// Block 256 thr = 8 waves; block tile 256(M) x 64(N); wave tile 32 x 64.
// W slab (shared by all waves) double-buffered in LDS.
// ---------------------------------------------------------------------------
__global__ __launch_bounds__(256)
void proj_bf16_kernel(const float* __restrict__ A, const float* __restrict__ W,
                      __bf16* __restrict__ C, int M, int N, int K) {
    __shared__ __align__(16) __bf16 wslab[2][64 * 32];   // 2 x 4 KB

    const int tid   = threadIdx.x;
    const int lane  = tid & 31;
    const int wave  = tid >> 5;
    const int mbase = blockIdx.y * 256 + wave * 32;
    const int nbase = blockIdx.x * 64;

    v8f acc[2][4] = {};

    stage_slab_f32(W, K, nbase, 0, wslab[0], tid);
    __syncthreads();

    for (int kb = 0; kb < K; kb += 32) {
        const int cur = (kb >> 5) & 1;
        if (kb + 32 < K)
            stage_slab_f32(W, K, nbase, kb + 32, wslab[cur ^ 1], tid);

        v16bf a0 = load_a_frag_f32(A, K, mbase,      kb, lane);
        v16bf a1 = load_a_frag_f32(A, K, mbase + 16, kb, lane);
#pragma unroll
        for (int t = 0; t < 4; ++t) {
            v16bf b = load_b_frag_lds(wslab[cur], t, lane);
            acc[0][t] = WMMA_BF16(a0, b, acc[0][t]);
            acc[1][t] = WMMA_BF16(a1, b, acc[1][t]);
        }
        __syncthreads();
    }

    const int row_off = (lane >> 4) * 8;
    const int col     = lane & 15;
#pragma unroll
    for (int f = 0; f < 2; ++f)
#pragma unroll
        for (int t = 0; t < 4; ++t)
#pragma unroll
            for (int r = 0; r < 8; ++r) {
                int m = mbase + f * 16 + r + row_off;
                int n = nbase + 16 * t + col;
                C[(size_t)m * N + n] = (__bf16)acc[f][t][r];
            }
}

// ---------------------------------------------------------------------------
// Stage 1b: gate weights w[row,h] = q[row,:] . Wg[h,:], fp32. Wave per row.
// ---------------------------------------------------------------------------
__global__ __launch_bounds__(256)
void gate_kernel(const float* __restrict__ q, const float* __restrict__ Wg,
                 float* __restrict__ w, int rows, int K) {
    const int lane = threadIdx.x & 31;
    const int wave = threadIdx.x >> 5;
    const int row  = blockIdx.x * 8 + wave;
    if (row >= rows) return;

    const float* qr = q + (size_t)row * K;
    float acc[NHEADS] = {0.f, 0.f, 0.f, 0.f};
    for (int k = lane; k < K; k += 32) {
        float qv = qr[k];
#pragma unroll
        for (int h = 0; h < NHEADS; ++h) acc[h] += qv * Wg[(size_t)h * K + k];
    }
#pragma unroll
    for (int h = 0; h < NHEADS; ++h) {
        float v = acc[h];
#pragma unroll
        for (int off = 16; off > 0; off >>= 1) v += __shfl_down(v, off, 32);
        if (lane == 0) w[(size_t)row * NHEADS + h] = v;
    }
}

// ---------------------------------------------------------------------------
// Stage 2: scores[b,q,k] = sum_h w[b,q,h] * relu(qi[b,q,h,:] . ki[b,k,h,:])
// qi: [B*Lq, 2048] bf16, ki: [B*Lc, 2048] bf16 (head h = cols h*512..+511).
// Block 256 thr = 8 waves; block tile 256(q) x 64(k); wave tile 32 x 64.
// ki slab (shared by all waves) double-buffered in LDS.
// ---------------------------------------------------------------------------
__global__ __launch_bounds__(256)
void score_kernel(const __bf16* __restrict__ qi, const __bf16* __restrict__ ki,
                  const float* __restrict__ w, float* __restrict__ out) {
    __shared__ __align__(16) __bf16 kslab[2][64 * 32];   // 2 x 4 KB

    const int tid   = threadIdx.x;
    const int lane  = tid & 31;
    const int wave  = tid >> 5;
    const int b     = blockIdx.z;
    const int qbase = blockIdx.y * 256 + wave * 32;      // q row within batch
    const int kbase = blockIdx.x * 64;                   // key col within batch

    const __bf16* qb = qi + (size_t)b * LQ * DMODEL;
    const __bf16* kp = ki + (size_t)b * LC * DMODEL;
    const float*  wb = w  + (size_t)b * LQ * NHEADS;

    const int row_off = (lane >> 4) * 8;
    const int col     = lane & 15;

    v8f score[2][4] = {};

#pragma unroll 1
    for (int h = 0; h < NHEADS; ++h) {
        const int dbase = h * HEADDIM;
        v8f acc[2][4] = {};

        stage_slab_bf(kp, DMODEL, kbase, dbase, kslab[0], tid);
        __syncthreads();

        for (int d = 0; d < HEADDIM; d += 32) {
            const int cur = (d >> 5) & 1;
            if (d + 32 < HEADDIM)
                stage_slab_bf(kp, DMODEL, kbase, dbase + d + 32, kslab[cur ^ 1], tid);

            v16bf a0 = load_a_frag_bf(qb, DMODEL, qbase,      dbase + d, lane);
            v16bf a1 = load_a_frag_bf(qb, DMODEL, qbase + 16, dbase + d, lane);
#pragma unroll
            for (int t = 0; t < 4; ++t) {
                v16bf bf = load_b_frag_lds(kslab[cur], t, lane);
                acc[0][t] = WMMA_BF16(a0, bf, acc[0][t]);
                acc[1][t] = WMMA_BF16(a1, bf, acc[1][t]);
            }
            __syncthreads();
        }

        // relu + per-row gate weight, fold into score accumulators
#pragma unroll
        for (int f = 0; f < 2; ++f) {
            float wv[8];
#pragma unroll
            for (int r = 0; r < 8; ++r)
                wv[r] = wb[(size_t)(qbase + f * 16 + r + row_off) * NHEADS + h];
#pragma unroll
            for (int t = 0; t < 4; ++t)
#pragma unroll
                for (int r = 0; r < 8; ++r) {
                    float v = acc[f][t][r];
                    score[f][t][r] += (v > 0.f ? v : 0.f) * wv[r];
                }
        }
    }

#pragma unroll
    for (int f = 0; f < 2; ++f)
#pragma unroll
        for (int t = 0; t < 4; ++t)
#pragma unroll
            for (int r = 0; r < 8; ++r) {
                int m = qbase + f * 16 + r + row_off;
                int n = kbase + 16 * t + col;
                out[((size_t)b * LQ + m) * LC + n] = score[f][t][r];
            }
}

// ---------------------------------------------------------------------------
extern "C" void kernel_launch(void* const* d_in, const int* in_sizes, int n_in,
                              void* d_out, int out_size, void* d_ws, size_t ws_size,
                              hipStream_t stream) {
    (void)in_sizes; (void)n_in; (void)out_size; (void)ws_size;

    const float* q  = (const float*)d_in[0];   // [B, Lq, D]
    const float* k  = (const float*)d_in[1];   // [B, Lc, D]
    const float* Wq = (const float*)d_in[2];   // [D, D]
    const float* Wk = (const float*)d_in[3];   // [D, D]
    const float* Wg = (const float*)d_in[4];   // [H, D]
    float* out = (float*)d_out;                // [B, Lq, Lc]

    // Workspace layout
    const size_t qi_bytes = (size_t)BATCH * LQ * DMODEL * sizeof(__bf16); // 33.5 MB
    const size_t ki_bytes = (size_t)BATCH * LC * DMODEL * sizeof(__bf16); //  8.4 MB
    __bf16* qi = (__bf16*)d_ws;
    __bf16* ki = (__bf16*)((char*)d_ws + qi_bytes);
    float*  wg = (float*)((char*)d_ws + qi_bytes + ki_bytes);             // 128 KB

    const int Mq = BATCH * LQ;  // 8192
    const int Mk = BATCH * LC;  // 2048

    // qi = q @ Wq^T  -> bf16
    proj_bf16_kernel<<<dim3(DMODEL / 64, Mq / 256), 256, 0, stream>>>(q, Wq, qi, Mq, DMODEL, DMODEL);
    // ki = k @ Wk^T  -> bf16
    proj_bf16_kernel<<<dim3(DMODEL / 64, Mk / 256), 256, 0, stream>>>(k, Wk, ki, Mk, DMODEL, DMODEL);
    // w = q @ Wg^T   -> fp32
    gate_kernel<<<dim3(Mq / 8), 256, 0, stream>>>(q, Wg, wg, Mq, DMODEL);
    // scores
    score_kernel<<<dim3(LC / 64, LQ / 256, BATCH), 256, 0, stream>>>(qi, ki, wg, out);
}
